// Encoder_80109730005312
// MI455X (gfx1250) — matrix operand order repaired and checked by reading, not compiled, not run on hardware
//
#include <hip/hip_runtime.h>
#include <hip/hip_bf16.h>

typedef __bf16 bf16;
typedef __attribute__((ext_vector_type(16))) __bf16 v16bf;
typedef __attribute__((ext_vector_type(8)))  __bf16 v8bf;
typedef __attribute__((ext_vector_type(8)))  float  v8f;
typedef int v4i_ __attribute__((vector_size(4 * sizeof(int))));

#define B_  4
#define T_  1024
#define E_  1024
#define H_  16
#define DH_ 64
#define SCALE_ 0.03125f   // T^-0.5 = 1024^-0.5

#ifndef __has_builtin
#define __has_builtin(x) 0
#endif

#if __has_builtin(__builtin_amdgcn_global_load_async_to_lds_b128)
#define HAVE_ASYNC_LDS 1
#else
#define HAVE_ASYNC_LDS 0
#endif

#if HAVE_ASYNC_LDS
static __device__ __forceinline__ void async_copy_b128(const void* g, void* l) {
    __builtin_amdgcn_global_load_async_to_lds_b128(
        (__attribute__((address_space(1))) v4i_*)(g),
        (__attribute__((address_space(3))) v4i_*)(l),
        0, 0);
}
static __device__ __forceinline__ void wait_async0() {
#if __has_builtin(__builtin_amdgcn_s_wait_asynccnt)
    __builtin_amdgcn_s_wait_asynccnt(0);
#else
    asm volatile("s_wait_asynccnt 0x0" ::: "memory");
#endif
}
#endif

union Frag { v16bf v; v8bf h[2]; };

static __device__ __forceinline__ v8f vzero8() {
    v8f z;
#pragma unroll
    for (int i = 0; i < 8; ++i) z[i] = 0.0f;
    return z;
}

static __device__ __forceinline__ v8f wmma_bf16(const Frag& a, const Frag& b, v8f c) {
    return __builtin_amdgcn_wmma_f32_16x16x32_bf16(
        /*neg_a=*/false, a.v, /*neg_b=*/false, b.v,
        /*c_mod=*/(short)0, c, /*reuse_a=*/false, /*reuse_b=*/false);
}

// ---------------------------------------------------------------------------
// Weight packing: wq/wk/wv [H,E,DH] fp32 -> Wqkv bf16 [K=E][N=3*E] row-major
// ---------------------------------------------------------------------------
__global__ __launch_bounds__(256)
void pack_qkv_w_kernel(const float* __restrict__ wq, const float* __restrict__ wk,
                       const float* __restrict__ wv, bf16* __restrict__ Wout) {
    int idx = blockIdx.x * 256 + threadIdx.x;      // total E_*3*E_
    int k = idx / (3 * E_);
    int n = idx - k * (3 * E_);
    int sel = n >> 10, rem = n & (E_ - 1);
    int h = rem >> 6, d = rem & (DH_ - 1);
    const float* src = (sel == 0) ? wq : (sel == 1) ? wk : wv;
    Wout[(size_t)k * (3 * E_) + n] = (bf16)src[((size_t)h * E_ + k) * DH_ + d];
}

__global__ __launch_bounds__(256)
void pack_qkv_bias_kernel(const float* __restrict__ bq, const float* __restrict__ bk,
                          const float* __restrict__ bv, float* __restrict__ bout) {
    int n = blockIdx.x * 256 + threadIdx.x;
    if (n >= 3 * E_) return;
    int sel = n >> 10, rem = n & (E_ - 1);
    const float* s = (sel == 0) ? bq : (sel == 1) ? bk : bv;
    bout[n] = s[rem];                               // rem == h*DH + d
}

__global__ __launch_bounds__(256)
void pack_f32_to_bf16_kernel(const float* __restrict__ in, bf16* __restrict__ out) {
    size_t i = (size_t)blockIdx.x * 256 + threadIdx.x;
    out[i] = (bf16)in[i];
}

// ---------------------------------------------------------------------------
// LayerNorm over rows of length E_ (4096 rows). Optional residual add,
// optional fp32 and bf16 outputs.
// ---------------------------------------------------------------------------
__global__ __launch_bounds__(256)
void ln_kernel(const float* __restrict__ in, const float* __restrict__ resid,
               const float* __restrict__ g, const float* __restrict__ bta,
               float* __restrict__ outF, bf16* __restrict__ outBF) {
    __shared__ float red[256];
    __shared__ float mu_s, rstd_s;
    const int row = blockIdx.x;
    const int tid = threadIdx.x;
    const size_t base = (size_t)row * E_;

    float v[4];
    float s = 0.f, ss = 0.f;
#pragma unroll
    for (int i = 0; i < 4; ++i) {
        int c = tid + i * 256;
        float t = in[base + c];
        if (resid) t += resid[base + c];
        v[i] = t; s += t; ss += t * t;
    }
    red[tid] = s; __syncthreads();
    for (int st = 128; st > 0; st >>= 1) {
        if (tid < st) red[tid] += red[tid + st];
        __syncthreads();
    }
    if (tid == 0) mu_s = red[0] * (1.0f / E_);
    __syncthreads();
    red[tid] = ss; __syncthreads();
    for (int st = 128; st > 0; st >>= 1) {
        if (tid < st) red[tid] += red[tid + st];
        __syncthreads();
    }
    if (tid == 0) {
        float var = red[0] * (1.0f / E_) - mu_s * mu_s;
        rstd_s = rsqrtf(var + 1e-5f);
    }
    __syncthreads();
    const float mu = mu_s, rstd = rstd_s;
#pragma unroll
    for (int i = 0; i < 4; ++i) {
        int c = tid + i * 256;
        float y = (v[i] - mu) * rstd * g[c] + bta[c];
        if (outF)  outF[base + c] = y;
        if (outBF) outBF[base + c] = (bf16)y;
    }
}

// ---------------------------------------------------------------------------
// Tiled bf16 WMMA GEMM, double-buffered LDS, async global->LDS when available.
//   C[M,N] = A[M,K] * Bm[K,N] + bias
//   mode 0: bf16 scatter into qkv buffer [3][B][H][T][DH]
//   mode 1: fp32 out = acc + bias[n] + resid[m*N+n]
// Block = 256 threads (8 waves), tile 128x128, k-chunk 32.
// ---------------------------------------------------------------------------
__global__ __launch_bounds__(256)
void gemm_bf16_kernel(const bf16* __restrict__ A, const bf16* __restrict__ Bm,
                      const float* __restrict__ bias, const float* __restrict__ resid,
                      float* __restrict__ outF, bf16* __restrict__ outQKV,
                      int M, int N, int K, int mode) {
    __shared__ bf16 As[2][128][40];   // [m][k], row stride 80B (16B aligned)
    __shared__ bf16 Bs[2][128][40];   // transposed: [n][k]

    const int tid  = threadIdx.x;
    const int lane = tid & 31;
    const int wid  = tid >> 5;
    const int wm   = wid >> 2;         // 0..1  (m dimension of wave grid)
    const int wn   = wid & 3;          // 0..3  (n dimension)
    const int col  = lane & 15;
    const int half = lane >> 4;
    const int m0 = blockIdx.y * 128;
    const int n0 = blockIdx.x * 128;

    // staging coordinates
    const int ar = tid >> 1, apart = tid & 1;   // A: row, 16-elem half
    const int sk = tid >> 3, sg = tid & 7;      // B: k within chunk, 16-col group

    v8f acc[4][2];
#pragma unroll
    for (int i = 0; i < 4; ++i)
#pragma unroll
        for (int j = 0; j < 2; ++j) acc[i][j] = vzero8();

    const bf16* aBase = A + (size_t)(m0 + ar) * K + apart * 16;
    const bf16* bBase = Bm + (size_t)sk * N + n0 + sg * 16;

    // ---- prologue: stage chunk 0 into buffer 0 ----
#if HAVE_ASYNC_LDS
    async_copy_b128(aBase,     &As[0][ar][apart * 16]);
    async_copy_b128(aBase + 8, &As[0][ar][apart * 16 + 8]);
#else
    {
        const v8bf* gp = (const v8bf*)aBase;
        v8bf a0 = gp[0], a1 = gp[1];
        *(v8bf*)&As[0][ar][apart * 16]     = a0;
        *(v8bf*)&As[0][ar][apart * 16 + 8] = a1;
    }
#endif
    {
        const v8bf* gp = (const v8bf*)bBase;
        v8bf r0 = gp[0], r1 = gp[1];
#pragma unroll
        for (int i = 0; i < 8; ++i) Bs[0][sg * 16 + i][sk]     = r0[i];
#pragma unroll
        for (int i = 0; i < 8; ++i) Bs[0][sg * 16 + 8 + i][sk] = r1[i];
    }
#if HAVE_ASYNC_LDS
    wait_async0();
#endif
    __syncthreads();

    const int kChunks = K >> 5;
    int buf = 0;
    v8bf bReg0, bReg1;
#if !HAVE_ASYNC_LDS
    v8bf aReg0, aReg1;
#endif

    for (int kc = 0; kc < kChunks; ++kc) {
        const int nb = buf ^ 1;
        const bool more = (kc + 1) < kChunks;

        // issue next-chunk global loads before computing (overlap with WMMA)
        if (more) {
            const bf16* an = aBase + (size_t)(kc + 1) * 32;
            const bf16* bn = bBase + (size_t)(kc + 1) * 32 * N;
#if HAVE_ASYNC_LDS
            async_copy_b128(an,     &As[nb][ar][apart * 16]);
            async_copy_b128(an + 8, &As[nb][ar][apart * 16 + 8]);
#else
            const v8bf* ga = (const v8bf*)an;
            aReg0 = ga[0]; aReg1 = ga[1];
#endif
            const v8bf* gb = (const v8bf*)bn;
            bReg0 = gb[0]; bReg1 = gb[1];
        }

        // ---- fragments + 8x WMMA from current buffer ----
        Frag afr[4], bfr[2];
#pragma unroll
        for (int mt = 0; mt < 4; ++mt) {
            int r = wm * 64 + mt * 16 + col;
            afr[mt].h[0] = *(const v8bf*)&As[buf][r][half * 16];
            afr[mt].h[1] = *(const v8bf*)&As[buf][r][half * 16 + 8];
        }
#pragma unroll
        for (int nt = 0; nt < 2; ++nt) {
            int n = wn * 32 + nt * 16 + col;
            bfr[nt].h[0] = *(const v8bf*)&Bs[buf][n][half * 16];
            bfr[nt].h[1] = *(const v8bf*)&Bs[buf][n][half * 16 + 8];
        }
#pragma unroll
        for (int mt = 0; mt < 4; ++mt)
#pragma unroll
            for (int nt = 0; nt < 2; ++nt)
                acc[mt][nt] = wmma_bf16(afr[mt], bfr[nt], acc[mt][nt]);

        // ---- publish next buffer ----
        if (more) {
#if !HAVE_ASYNC_LDS
            *(v8bf*)&As[nb][ar][apart * 16]     = aReg0;
            *(v8bf*)&As[nb][ar][apart * 16 + 8] = aReg1;
#endif
#pragma unroll
            for (int i = 0; i < 8; ++i) Bs[nb][sg * 16 + i][sk]     = bReg0[i];
#pragma unroll
            for (int i = 0; i < 8; ++i) Bs[nb][sg * 16 + 8 + i][sk] = bReg1[i];
#if HAVE_ASYNC_LDS
            wait_async0();
#endif
        }
        __syncthreads();
        buf = nb;
    }

    // ---- epilogue: hoisted per-tile addressing ----
#pragma unroll
    for (int mt = 0; mt < 4; ++mt) {
#pragma unroll
        for (int nt = 0; nt < 2; ++nt) {
            const int mBase = m0 + wm * 64 + mt * 16 + half * 8;
            const int n = n0 + wn * 32 + nt * 16 + col;
            const float bn = bias[n];
            if (mode == 0) {
                const int sel = n >> 10, rem = n & (E_ - 1);
                const int h = rem >> 6, d = rem & (DH_ - 1);
                const int b = mBase >> 10, t = mBase & (T_ - 1);
                bf16* p = outQKV +
                    (((size_t)sel * (B_ * H_) + b * H_ + h) * T_ + t) * DH_ + d;
#pragma unroll
                for (int r = 0; r < 8; ++r)
                    p[(size_t)r * DH_] = (bf16)(acc[mt][nt][r] + bn);
            } else {
                float* p = outF + (size_t)mBase * N + n;
                const float* q = resid + (size_t)mBase * N + n;
#pragma unroll
                for (int r = 0; r < 8; ++r)
                    p[(size_t)r * N] = acc[mt][nt][r] + bn + q[(size_t)r * N];
            }
        }
    }
}

// ---------------------------------------------------------------------------
// Flash attention: grid = B*H*(T/64) blocks, 128 threads (4 waves).
// Each wave owns 16 query rows; iterate 16 key/value tiles of 64.
// ---------------------------------------------------------------------------
__global__ __launch_bounds__(128)
void flash_attn_kernel(const bf16* __restrict__ qkv, float* __restrict__ ctx) {
    __shared__ bf16 Qs[64][72];
    __shared__ bf16 Ks[64][72];        // [key s][d]
    __shared__ bf16 Vts[64][72];       // transposed: [d][s]
    __shared__ bf16 Ps[4][16][72];     // per-wave P tile [m][s]

    const int tid  = threadIdx.x;
    const int lane = tid & 31;
    const int w    = tid >> 5;
    const int col  = lane & 15;
    const int half = lane >> 4;
    const int qb = blockIdx.x & 15;    // query block (T/64 = 16)
    const int bh = blockIdx.x >> 4;    // b*H + h
    const int b  = bh >> 4, h = bh & (H_ - 1);
    const size_t BHT = (size_t)B_ * H_ * T_ * DH_;
    const bf16* Q  = qkv + (size_t)bh * T_ * DH_;
    const bf16* Kg = qkv + BHT + (size_t)bh * T_ * DH_;
    const bf16* Vg = qkv + 2 * BHT + (size_t)bh * T_ * DH_;

    const int sr = tid >> 1, spart = tid & 1;   // staging coords

    // stage Q tile [64 x 64]
    {
        const v8bf* gp = (const v8bf*)(Q + (size_t)(qb * 64 + sr) * DH_ + spart * 32);
        v8bf x0 = gp[0], x1 = gp[1], x2 = gp[2], x3 = gp[3];
        *(v8bf*)&Qs[sr][spart * 32]      = x0;
        *(v8bf*)&Qs[sr][spart * 32 + 8]  = x1;
        *(v8bf*)&Qs[sr][spart * 32 + 16] = x2;
        *(v8bf*)&Qs[sr][spart * 32 + 24] = x3;
    }
    __syncthreads();
    Frag aq[2];
#pragma unroll
    for (int c = 0; c < 2; ++c) {
        aq[c].h[0] = *(const v8bf*)&Qs[w * 16 + col][c * 32 + half * 16];
        aq[c].h[1] = *(const v8bf*)&Qs[w * 16 + col][c * 32 + half * 16 + 8];
    }

    float mrow[8], lrow[8];
    v8f o[4];
#pragma unroll
    for (int r = 0; r < 8; ++r) { mrow[r] = -3.0e38f; lrow[r] = 0.f; }
#pragma unroll
    for (int j = 0; j < 4; ++j) o[j] = vzero8();

    for (int kt = 0; kt < 16; ++kt) {
        __syncthreads();   // previous iteration's consumers of Ks/Vts done
        // stage K tile [64 x 64] (async when available)
        {
            const bf16* gp = Kg + (size_t)(kt * 64 + sr) * DH_ + spart * 32;
#if HAVE_ASYNC_LDS
            async_copy_b128(gp,      &Ks[sr][spart * 32]);
            async_copy_b128(gp + 8,  &Ks[sr][spart * 32 + 8]);
            async_copy_b128(gp + 16, &Ks[sr][spart * 32 + 16]);
            async_copy_b128(gp + 24, &Ks[sr][spart * 32 + 24]);
#else
            const v8bf* g8 = (const v8bf*)gp;
            v8bf x0 = g8[0], x1 = g8[1], x2 = g8[2], x3 = g8[3];
            *(v8bf*)&Ks[sr][spart * 32]      = x0;
            *(v8bf*)&Ks[sr][spart * 32 + 8]  = x1;
            *(v8bf*)&Ks[sr][spart * 32 + 16] = x2;
            *(v8bf*)&Ks[sr][spart * 32 + 24] = x3;
#endif
        }
        // stage V tile transposed: Vts[d][s] (scalar scatter)
        {
            const bf16* gp = Vg + (size_t)(kt * 64 + sr) * DH_ + spart * 32;
#pragma unroll
            for (int i = 0; i < 32; ++i) Vts[spart * 32 + i][sr] = gp[i];
        }
        // prefetch next K/V tiles into cache
        if (kt + 1 < 16) {
            __builtin_prefetch(Kg + (size_t)((kt + 1) * 64 + sr) * DH_ + spart * 32, 0, 1);
            __builtin_prefetch(Vg + (size_t)((kt + 1) * 64 + sr) * DH_ + spart * 32, 0, 1);
        }
#if HAVE_ASYNC_LDS
        wait_async0();
#endif
        __syncthreads();

        // S = Q K^T  (per wave: 16 x 64, 4 n-tiles, 2 k-chunks)
        v8f s[4];
#pragma unroll
        for (int j = 0; j < 4; ++j) {
            s[j] = vzero8();
#pragma unroll
            for (int c = 0; c < 2; ++c) {
                Frag bk;
                bk.h[0] = *(const v8bf*)&Ks[j * 16 + col][c * 32 + half * 16];
                bk.h[1] = *(const v8bf*)&Ks[j * 16 + col][c * 32 + half * 16 + 8];
                s[j] = wmma_bf16(aq[c], bk, s[j]);
            }
        }
#pragma unroll
        for (int j = 0; j < 4; ++j)
#pragma unroll
            for (int r = 0; r < 8; ++r) s[j][r] *= SCALE_;

        // online softmax per row (row m = half*8 + r; columns across 16 lanes)
#pragma unroll
        for (int r = 0; r < 8; ++r) {
            float mx = s[0][r];
#pragma unroll
            for (int j = 1; j < 4; ++j) mx = fmaxf(mx, s[j][r]);
#pragma unroll
            for (int d = 8; d >= 1; d >>= 1) mx = fmaxf(mx, __shfl_xor(mx, d, 32));
            float mnew  = fmaxf(mrow[r], mx);
            float alpha = __expf(mrow[r] - mnew);
            float ssum  = 0.f;
#pragma unroll
            for (int j = 0; j < 4; ++j) {
                float p = __expf(s[j][r] - mnew);
                s[j][r] = p; ssum += p;
            }
#pragma unroll
            for (int d = 8; d >= 1; d >>= 1) ssum += __shfl_xor(ssum, d, 32);
            lrow[r] = lrow[r] * alpha + ssum;
            mrow[r] = mnew;
#pragma unroll
            for (int j = 0; j < 4; ++j) o[j][r] *= alpha;
        }

        // write P (C layout -> LDS row-major [m][s])
#pragma unroll
        for (int j = 0; j < 4; ++j)
#pragma unroll
            for (int r = 0; r < 8; ++r)
                Ps[w][half * 8 + r][j * 16 + col] = (bf16)s[j][r];
        __syncthreads();

        // O += P V  (4 d-tiles, 2 s-chunks)
        Frag ap[2];
#pragma unroll
        for (int c = 0; c < 2; ++c) {
            ap[c].h[0] = *(const v8bf*)&Ps[w][col][c * 32 + half * 16];
            ap[c].h[1] = *(const v8bf*)&Ps[w][col][c * 32 + half * 16 + 8];
        }
#pragma unroll
        for (int j = 0; j < 4; ++j) {
#pragma unroll
            for (int c = 0; c < 2; ++c) {
                Frag bv;
                bv.h[0] = *(const v8bf*)&Vts[j * 16 + col][c * 32 + half * 16];
                bv.h[1] = *(const v8bf*)&Vts[j * 16 + col][c * 32 + half * 16 + 8];
                o[j] = wmma_bf16(ap[c], bv, o[j]);
            }
        }
    }

    // finalize: O /= l, scatter ctx as fp32 [B,T,E] with column h*DH+d
#pragma unroll
    for (int r = 0; r < 8; ++r) {
        float inv = 1.0f / lrow[r];
        int t = qb * 64 + w * 16 + half * 8 + r;
#pragma unroll
        for (int j = 0; j < 4; ++j) {
            int d = j * 16 + col;
            ctx[((size_t)b * T_ + t) * E_ + h * DH_ + d] = o[j][r] * inv;
        }
    }
}

// ---------------------------------------------------------------------------
// Last-head attention map: softmax(q_{H-1} k_{H-1}^T * scale) -> [B,T,T]
// One block per (b,t), 256 threads, 4 keys per thread.
// ---------------------------------------------------------------------------
__global__ __launch_bounds__(256)
void attn_map_kernel(const bf16* __restrict__ qkv, float* __restrict__ out2) {
    __shared__ float qs[64];
    __shared__ float sc[1024];
    __shared__ float red[256];
    __shared__ float stat;
    const int bt = blockIdx.x;
    const int b = bt >> 10, t = bt & (T_ - 1);
    const int tid = threadIdx.x;
    const size_t BHT = (size_t)B_ * H_ * T_ * DH_;
    const int bh = b * H_ + (H_ - 1);
    const bf16* Q  = qkv + (size_t)bh * T_ * DH_ + (size_t)t * DH_;
    const bf16* Kg = qkv + BHT + (size_t)bh * T_ * DH_;

    if (tid < 64) qs[tid] = (float)Q[tid];
    __syncthreads();

    float lmax = -3.0e38f;
#pragma unroll
    for (int kk = 0; kk < 4; ++kk) {
        int s_ = tid + kk * 256;
        const bf16* kr = Kg + (size_t)s_ * DH_;
        float d = 0.f;
#pragma unroll
        for (int i = 0; i < 64; ++i) d += qs[i] * (float)kr[i];
        d *= SCALE_;
        sc[s_] = d;
        lmax = fmaxf(lmax, d);
    }
    red[tid] = lmax; __syncthreads();
    for (int st = 128; st > 0; st >>= 1) {
        if (tid < st) red[tid] = fmaxf(red[tid], red[tid + st]);
        __syncthreads();
    }
    if (tid == 0) stat = red[0];
    __syncthreads();
    const float mx = stat;

    float lsum = 0.f;
#pragma unroll
    for (int kk = 0; kk < 4; ++kk) {
        int s_ = tid + kk * 256;
        float e = __expf(sc[s_] - mx);
        sc[s_] = e; lsum += e;
    }
    red[tid] = lsum; __syncthreads();
    for (int st = 128; st > 0; st >>= 1) {
        if (tid < st) red[tid] += red[tid + st];
        __syncthreads();
    }
    if (tid == 0) stat = red[0];
    __syncthreads();
    const float inv = 1.0f / stat;
#pragma unroll
    for (int kk = 0; kk < 4; ++kk) {
        int s_ = tid + kk * 256;
        out2[((size_t)b * T_ + t) * T_ + s_] = sc[s_] * inv;
    }
}

// ---------------------------------------------------------------------------
extern "C" void kernel_launch(void* const* d_in, const int* in_sizes, int n_in,
                              void* d_out, int out_size, void* d_ws, size_t ws_size,
                              hipStream_t stream) {
    (void)in_sizes; (void)n_in; (void)out_size; (void)ws_size;
    const float* x     = (const float*)d_in[0];
    const float* wq    = (const float*)d_in[1];
    const float* bq    = (const float*)d_in[2];
    const float* wk    = (const float*)d_in[3];
    const float* bk    = (const float*)d_in[4];
    const float* wv    = (const float*)d_in[5];
    const float* bv    = (const float*)d_in[6];
    const float* ffw   = (const float*)d_in[7];
    const float* ffb   = (const float*)d_in[8];
    const float* ln1_g = (const float*)d_in[9];
    const float* ln1_b = (const float*)d_in[10];
    const float* ln2_g = (const float*)d_in[11];
    const float* ln2_b = (const float*)d_in[12];
    const float* ln3_g = (const float*)d_in[13];
    const float* ln3_b = (const float*)d_in[14];

    float* outLN   = (float*)d_out;                                   // [B,T,E]
    float* outAttn = (float*)d_out + (size_t)B_ * T_ * E_;            // [B,T,T]

    const int R = B_ * T_;                        // 4096 rows
    char* ws = (char*)d_ws;
    size_t off = 0;
    auto alloc = [&](size_t bytes) { char* p = ws + off; off += (bytes + 255) & ~(size_t)255; return p; };
    bf16*  x1bf    = (bf16*) alloc((size_t)R * E_ * 2);               // 8 MB
    bf16*  Wqkv    = (bf16*) alloc((size_t)E_ * 3 * E_ * 2);          // 6 MB
    float* biasQKV = (float*)alloc((size_t)3 * E_ * 4);               // 12 KB
    bf16*  qkv     = (bf16*) alloc((size_t)3 * R * E_ * 2);           // 24 MB
    float* ctx     = (float*)alloc((size_t)R * E_ * 4);               // 16 MB
    float* x2f     = (float*)alloc((size_t)R * E_ * 4);               // 16 MB
    bf16*  x2bf    = (bf16*) alloc((size_t)R * E_ * 2);               // 8 MB
    bf16*  ffwbf   = (bf16*) alloc((size_t)E_ * E_ * 2);              // 2 MB
    float* yf      = (float*)alloc((size_t)R * E_ * 4);               // 16 MB

    // 1. pack weights to bf16
    pack_qkv_w_kernel<<<(E_ * 3 * E_) / 256, 256, 0, stream>>>(wq, wk, wv, Wqkv);
    pack_qkv_bias_kernel<<<(3 * E_ + 255) / 256, 256, 0, stream>>>(bq, bk, bv, biasQKV);
    pack_f32_to_bf16_kernel<<<(E_ * E_) / 256, 256, 0, stream>>>(ffw, ffwbf);

    // 2. LN1 -> bf16 activations
    ln_kernel<<<R, 256, 0, stream>>>(x, nullptr, ln1_g, ln1_b, nullptr, x1bf);

    // 3. fused QKV projection: [4096,1024] x [1024,3072] -> qkv bf16 [3,B,H,T,DH]
    gemm_bf16_kernel<<<dim3((3 * E_) / 128, R / 128), 256, 0, stream>>>(
        x1bf, Wqkv, biasQKV, nullptr, nullptr, qkv, R, 3 * E_, E_, 0);

    // 4. flash attention -> ctx fp32 [B,T,E]
    flash_attn_kernel<<<B_ * H_ * (T_ / 64), 128, 0, stream>>>(qkv, ctx);

    // 5. last-head attention map -> second output
    attn_map_kernel<<<B_ * T_, 256, 0, stream>>>(qkv, outAttn);

    // 6. x2 = LN2(x + ctx) -> fp32 (residual for FFN) + bf16 (GEMM input)
    ln_kernel<<<R, 256, 0, stream>>>(x, ctx, ln2_g, ln2_b, x2f, x2bf);

    // 7. y = x2 + x2 @ ffw + ffb
    gemm_bf16_kernel<<<dim3(E_ / 128, R / 128), 256, 0, stream>>>(
        x2bf, ffwbf, ffb, x2f, yf, nullptr, R, E_, E_, 1);

    // 8. out = LN3(y)
    ln_kernel<<<R, 256, 0, stream>>>(yf, nullptr, ln3_g, ln3_b, outLN, nullptr);
}
